// GATGRUCellInversed_36009005809887
// MI455X (gfx1250) — matrix operand order repaired and checked by reading, not compiled, not run on hardware
//
#include <hip/hip_runtime.h>
#include <math.h>
#include <stdint.h>

// ---------------------------------------------------------------------------
// GATGRUCellInversed for MI455X (gfx1250, wave32, WMMA).
// Dense projections run through v_wmma_f32_16x16x32_bf16 (bf16 A/B, f32 acc)
// with double-buffered LDS tiles filled by GLOBAL_LOAD_ASYNC_TO_LDS_B128
// (ASYNCcnt). Attention exploits the fixed in-degree-8 mailbox layout
// (dst == repeat(arange(NT), 8)).
// ---------------------------------------------------------------------------

typedef __attribute__((ext_vector_type(16))) __bf16 v16bf;
typedef __attribute__((ext_vector_type(8)))  float  v8f;
typedef __attribute__((vector_size(16)))     int    v4i_t;

#define NT_   32768
#define NF_   768
#define H_    8
#define HD_   96
#define DEG_  8
#define G3_   2304   // 3*768

#if defined(__has_builtin)
#if __has_builtin(__builtin_amdgcn_global_load_async_to_lds_b128)
#define HAVE_ASYNC_LDS 1
#endif
#endif
#ifndef HAVE_ASYNC_LDS
#define HAVE_ASYNC_LDS 0
#endif

#define AS1 __attribute__((address_space(1)))
#define AS3 __attribute__((address_space(3)))

__device__ __forceinline__ void async_copy16(const unsigned short* g, unsigned short* l) {
#if HAVE_ASYNC_LDS
    // per-lane 16B global -> LDS DMA, tracked with ASYNCcnt.
    // builtin signature: (v4i AS1* global_src, v4i AS3* lds_dst, imm off, imm cpol)
    __builtin_amdgcn_global_load_async_to_lds_b128(
        (AS1 v4i_t*)(uintptr_t)g,
        (AS3 v4i_t*)(unsigned int)(uintptr_t)l, 0, 0);
#else
    *(uint4*)l = *(const uint4*)g;     // sync fallback: global_load + ds_store
#endif
}

__device__ __forceinline__ void wait_async0() {
#if HAVE_ASYNC_LDS
#if __has_builtin(__builtin_amdgcn_s_wait_asynccnt)
    __builtin_amdgcn_s_wait_asynccnt(0);
#else
    asm volatile("s_wait_asynccnt 0x0" ::: "memory");
#endif
#endif
}

__device__ __forceinline__ unsigned short f2bf(float x) {
    unsigned int u = __float_as_uint(x);
    u += 0x7FFFu + ((u >> 16) & 1u);          // round-to-nearest-even
    return (unsigned short)(u >> 16);
}

// ---------------------------------------------------------------------------
// f32 -> bf16 flat convert
// ---------------------------------------------------------------------------
__global__ void f32_to_bf16_kernel(const float* __restrict__ in,
                                   unsigned short* __restrict__ out, int n) {
    int i = blockIdx.x * blockDim.x + threadIdx.x;
    if (i < n) out[i] = f2bf(in[i]);
}

// Wg/Wx [H][F][HD] f32  ->  BT[(h*HD+d)][f] bf16   (per-output-column K-major)
__global__ void pack_w_hfd_kernel(const float* __restrict__ W,
                                  unsigned short* __restrict__ BT) {
    int i = blockIdx.x * blockDim.x + threadIdx.x;   // over NF_*NF_
    if (i >= NF_ * NF_) return;
    int f = i % NF_;
    int n = i / NF_;            // n = h*HD + d
    int h = n / HD_, d = n % HD_;
    BT[i] = f2bf(W[((size_t)h * NF_ + f) * HD_ + d]);
}

// ---------------------------------------------------------------------------
// bf16 GEMM with f32 accumulate:  C[M][N] = A[M][K] * BT[N][K]^T + bias[N]
// Block tile 128(M) x 128(N), 256 threads = 8 wave32 in a 4x2 grid; each wave
// owns 32x64 of C = 2x4 WMMA 16x16 tiles (8 v_wmma per K-step of 32).
// LDS tiles are double-buffered; stage k+1 is fetched asynchronously while
// stage k runs WMMA, with s_wait_asynccnt placed after the math.
// ---------------------------------------------------------------------------
#define BM 128
#define BN 128
#define BK 32
#define LDST 40   // halves per LDS tile row (80 B: 16B-aligned, bank-skewed)

__global__ __launch_bounds__(256)
void gemm_bf16_kernel(const unsigned short* __restrict__ A,   // [M][K] bf16
                      const unsigned short* __restrict__ BT,  // [N][K] bf16
                      const float* __restrict__ bias,         // [N]
                      float* __restrict__ C,                  // [M][N] f32
                      int M, int N, int K) {
    __shared__ unsigned short As[2][BM * LDST];
    __shared__ unsigned short Bs[2][BN * LDST];

    const int tid   = threadIdx.x;
    const int lane  = tid & 31;
    const int wave  = tid >> 5;         // 0..7
    const int wm    = wave & 3;         // M sub-block (32 rows each)
    const int wn    = wave >> 2;        // N sub-block (64 cols each)
    const int lrow  = lane & 15;
    const bool hiL  = lane >= 16;
    const int blockM = blockIdx.x * BM;
    const int blockN = blockIdx.y * BN;

    // ---- staging chunk assignment: 512 16B chunks per tile, 2 per thread
    // chunk c: row = c>>2 (tile row), seg = c&3 (8-half segment)
    const int cA0 = tid, cA1 = tid + 256;
    const int rA0 = cA0 >> 2, sA0 = cA0 & 3;
    const int rA1 = cA1 >> 2, sA1 = cA1 & 3;
    const unsigned short* pA0 = A + (size_t)(blockM + rA0) * K + sA0 * 8;
    const unsigned short* pA1 = A + (size_t)(blockM + rA1) * K + sA1 * 8;
    const unsigned short* pB0 = BT + (size_t)(blockN + rA0) * K + sA0 * 8;
    const unsigned short* pB1 = BT + (size_t)(blockN + rA1) * K + sA1 * 8;
    const int lOffA0 = rA0 * LDST + sA0 * 8;
    const int lOffA1 = rA1 * LDST + sA1 * 8;

    v8f acc[2][4] = {};

    const int KT = K / BK;
    // prologue: stage 0 -> buffer 0
    async_copy16(pA0, &As[0][lOffA0]);
    async_copy16(pA1, &As[0][lOffA1]);
    async_copy16(pB0, &Bs[0][lOffA0]);
    async_copy16(pB1, &Bs[0][lOffA1]);
    pA0 += BK; pA1 += BK; pB0 += BK; pB1 += BK;
    wait_async0();
    __syncthreads();

    for (int kt = 0; kt < KT; ++kt) {
        const int cur = kt & 1;
        if (kt + 1 < KT) {          // prefetch next stage into other buffer
            const int nxt = cur ^ 1;
            async_copy16(pA0, &As[nxt][lOffA0]);
            async_copy16(pA1, &As[nxt][lOffA1]);
            async_copy16(pB0, &Bs[nxt][lOffA0]);
            async_copy16(pB1, &Bs[nxt][lOffA1]);
            pA0 += BK; pA1 += BK; pB0 += BK; pB1 += BK;
        }

        // ---- A fragments: 16x32 bf16, lanes 0-15 K{0..7,16..23}, 16-31 K{8..15,24..31}
        v16bf afrag[2];
#pragma unroll
        for (int i = 0; i < 2; ++i) {
            int r   = wm * 32 + i * 16 + lrow;
            int kof = hiL ? 8 : 0;
            union { uint4 u[2]; v16bf v; } u;
            u.u[0] = *(const uint4*)(&As[cur][r * LDST + kof]);
            u.u[1] = *(const uint4*)(&As[cur][r * LDST + kof + 16]);
            afrag[i] = u.v;
        }
        // ---- B fragments: 32x16 bf16, lane = column, lanes 0-15 K{0..7,16..23}, 16-31 K{8..15,24..31}
        v16bf bfrag[4];
#pragma unroll
        for (int j = 0; j < 4; ++j) {
            int cN  = wn * 64 + j * 16 + lrow;
            int kof = hiL ? 8 : 0;
            union { uint4 u[2]; v16bf v; } u;
            u.u[0] = *(const uint4*)(&Bs[cur][cN * LDST + kof]);
            u.u[1] = *(const uint4*)(&Bs[cur][cN * LDST + kof + 16]);
            bfrag[j] = u.v;
        }
#pragma unroll
        for (int i = 0; i < 2; ++i)
#pragma unroll
            for (int j = 0; j < 4; ++j)
                acc[i][j] = __builtin_amdgcn_wmma_f32_16x16x32_bf16(
                    false, afrag[i], false, bfrag[j],
                    (short)0, acc[i][j], false, false);

        if (kt + 1 < KT) wait_async0();   // wait hidden behind the 8 WMMAs
        __syncthreads();
    }

    // ---- epilogue: D layout = lanes 0-15 rows v, lanes 16-31 rows v+8
#pragma unroll
    for (int i = 0; i < 2; ++i) {
        int rbase = blockM + wm * 32 + i * 16 + (hiL ? 8 : 0);
#pragma unroll
        for (int j = 0; j < 4; ++j) {
            int col = blockN + wn * 64 + j * 16 + lrow;
            float b = bias ? bias[col] : 0.0f;
#pragma unroll
            for (int v = 0; v < 8; ++v)
                C[(size_t)(rbase + v) * N + col] = acc[i][j][v] + b;
        }
    }
}

// ---------------------------------------------------------------------------
// Per-(node,head) attention scores: P viewed as [N][H][HD]
//   sc_s[n][h] = sum_d P[n][h][d]*a[h][d];  sc_d[n][h] = sum_d P[n][h][d]*a[h][HD+d]
// ---------------------------------------------------------------------------
__global__ void scores_kernel(const float* __restrict__ P,
                              const float* __restrict__ a,    // [H][2*HD]
                              float* __restrict__ sc_s,
                              float* __restrict__ sc_d, int n_nodes) {
    int i = blockIdx.x * blockDim.x + threadIdx.x;    // n_nodes*H
    if (i >= n_nodes * H_) return;
    int n = i >> 3, h = i & 7;
    const float* p  = P + (size_t)n * NF_ + h * HD_;
    const float* av = a + h * (2 * HD_);
    float s0 = 0.f, s1 = 0.f;
#pragma unroll 4
    for (int d = 0; d < HD_; ++d) { float v = p[d]; s0 += v * av[d]; s1 += v * av[HD_ + d]; }
    sc_s[i] = s0; sc_d[i] = s1;
}

// ---------------------------------------------------------------------------
// Mailbox attention: one block per dst node; 8 contiguous edges per dst.
// LeakyReLU -> per-head softmax over 8 edges -> weighted sum of src rows.
// Optional accumulate (hid = 0.5 h_c + 0.5 h_s) and fused bf16 output.
// ---------------------------------------------------------------------------
__global__ __launch_bounds__(256)
void attend_kernel(const int* __restrict__ src,          // [NT*DEG]
                   const float* __restrict__ WhS,        // [Ns][NF] source feats
                   const float* __restrict__ sc_s,       // [Ns][H]
                   const float* __restrict__ sc_d,       // [NT][H]
                   const float* __restrict__ ab,         // [H]
                   float* __restrict__ outF,             // [NT][NF]
                   unsigned short* __restrict__ outB,    // bf16 mirror or null
                   float scale, int accumulate) {
    __shared__ float att[DEG_][H_];
    __shared__ int   sIdx[DEG_];
    const int t   = blockIdx.x;
    const int tid = threadIdx.x;

    if (tid < DEG_) sIdx[tid] = src[t * DEG_ + tid];
    __syncthreads();
    if (tid < DEG_ * H_) {
        int j = tid >> 3, hh = tid & 7;
        float e = sc_s[(size_t)sIdx[j] * H_ + hh] + sc_d[(size_t)t * H_ + hh] + ab[hh];
        att[j][hh] = (e > 0.f) ? e : 0.2f * e;     // LeakyReLU(0.2)
    }
    __syncthreads();
    if (tid < H_) {
        int hh = tid;
        float m = -3.4e38f;
#pragma unroll
        for (int j = 0; j < DEG_; ++j) m = fmaxf(m, att[j][hh]);
        float ex[DEG_]; float den = 0.f;
#pragma unroll
        for (int j = 0; j < DEG_; ++j) { ex[j] = __expf(att[j][hh] - m); den += ex[j]; }
        float inv = 1.0f / den;
#pragma unroll
        for (int j = 0; j < DEG_; ++j) att[j][hh] = ex[j] * inv;
    }
    __syncthreads();

    for (int f = tid; f < NF_; f += 256) {
        int hh = f / HD_;
        float acc = 0.f;
#pragma unroll
        for (int j = 0; j < DEG_; ++j)
            acc += att[j][hh] * WhS[(size_t)sIdx[j] * NF_ + f];
        size_t o = (size_t)t * NF_ + f;
        float r = scale * acc;
        if (accumulate) r += outF[o];
        outF[o] = r;
        if (outB) outB[o] = f2bf(r);
    }
}

// ---------------------------------------------------------------------------
// GRU cell elementwise epilogue
// ---------------------------------------------------------------------------
__global__ void gru_kernel(const float* __restrict__ gi,   // [NT][3*NF]
                           const float* __restrict__ gh,   // [NT][3*NF]
                           const float* __restrict__ hid,  // [NT][NF]
                           float* __restrict__ out) {
    int i = blockIdx.x * blockDim.x + threadIdx.x;          // NT*NF
    if (i >= NT_ * NF_) return;
    int n = i / NF_, c = i % NF_;
    const float* a = gi + (size_t)n * G3_;
    const float* b = gh + (size_t)n * G3_;
    float r    = 1.0f / (1.0f + __expf(-(a[c] + b[c])));
    float z    = 1.0f / (1.0f + __expf(-(a[NF_ + c] + b[NF_ + c])));
    float cand = tanhf(a[2 * NF_ + c] + r * b[2 * NF_ + c]);
    out[i] = (1.0f - z) * cand + z * hid[i];
}

// ---------------------------------------------------------------------------
extern "C" void kernel_launch(void* const* d_in, const int* in_sizes, int n_in,
                              void* d_out, int out_size, void* d_ws, size_t ws_size,
                              hipStream_t stream) {
    (void)in_sizes; (void)n_in; (void)out_size; (void)ws_size;
    const float* h    = (const float*)d_in[0];
    const float* hbc  = (const float*)d_in[1];
    const float* hbs  = (const float*)d_in[2];
    const float* Wg   = (const float*)d_in[3];
    const float* bg   = (const float*)d_in[4];
    const float* ag   = (const float*)d_in[5];
    const float* agb  = (const float*)d_in[6];
    const float* Wx   = (const float*)d_in[7];
    const float* bx   = (const float*)d_in[8];
    const float* ax   = (const float*)d_in[9];
    const float* axb  = (const float*)d_in[10];
    const float* Wih  = (const float*)d_in[11];
    const float* Whh  = (const float*)d_in[12];
    const float* bih  = (const float*)d_in[13];
    const float* bhh  = (const float*)d_in[14];
    const int* intra_src = (const int*)d_in[15];
    const int* c_src     = (const int*)d_in[17];
    const int* s_src     = (const int*)d_in[19];

    const size_t NB = (size_t)NT_ * NF_;
    char* ws = (char*)d_ws;
    size_t off = 0;
    auto alloc = [&](size_t bytes) -> void* {
        off = (off + 255) & ~(size_t)255;
        void* p = ws + off; off += bytes; return p;
    };

    unsigned short* hB   = (unsigned short*)alloc(NB * 2);
    unsigned short* hcB  = (unsigned short*)alloc(NB * 2);
    unsigned short* hsB  = (unsigned short*)alloc(NB * 2);
    unsigned short* xB   = (unsigned short*)alloc(NB * 2);
    unsigned short* hidB = (unsigned short*)alloc(NB * 2);
    unsigned short* BTg  = (unsigned short*)alloc((size_t)NF_ * NF_ * 2);
    unsigned short* BTx  = (unsigned short*)alloc((size_t)NF_ * NF_ * 2);
    unsigned short* BTih = (unsigned short*)alloc((size_t)G3_ * NF_ * 2);
    unsigned short* BThh = (unsigned short*)alloc((size_t)G3_ * NF_ * 2);
    float* Wh   = (float*)alloc(NB * 4);
    float* xF   = (float*)alloc(NB * 4);
    float* Wxd  = (float*)alloc(NB * 4);
    float* Wcs  = (float*)alloc(NB * 4);
    float* Wss  = (float*)alloc(NB * 4);
    float* hidF = (float*)alloc(NB * 4);
    float* gi   = (float*)alloc((size_t)NT_ * G3_ * 4);
    float* gh   = (float*)alloc((size_t)NT_ * G3_ * 4);
    float* scAs = (float*)alloc((size_t)NT_ * H_ * 4);
    float* scAd = (float*)alloc((size_t)NT_ * H_ * 4);
    float* scXs = (float*)alloc((size_t)NT_ * H_ * 4);  // dummy half
    float* scXd = (float*)alloc((size_t)NT_ * H_ * 4);
    float* scCs = (float*)alloc((size_t)NT_ * H_ * 4);
    float* scCd = (float*)alloc((size_t)NT_ * H_ * 4);  // dummy half
    float* scSs = (float*)alloc((size_t)NT_ * H_ * 4);
    float* scSd = (float*)alloc((size_t)NT_ * H_ * 4);  // dummy half

    const int T = 256;
    auto cdiv = [](size_t a, size_t b) { return (int)((a + b - 1) / b); };

    // ---- weight packing (bf16) ----
    pack_w_hfd_kernel<<<cdiv((size_t)NF_ * NF_, T), T, 0, stream>>>(Wg, BTg);
    pack_w_hfd_kernel<<<cdiv((size_t)NF_ * NF_, T), T, 0, stream>>>(Wx, BTx);
    f32_to_bf16_kernel<<<cdiv((size_t)G3_ * NF_, T), T, 0, stream>>>(Wih, BTih, G3_ * NF_);
    f32_to_bf16_kernel<<<cdiv((size_t)G3_ * NF_, T), T, 0, stream>>>(Whh, BThh, G3_ * NF_);
    // ---- feature conversion ----
    f32_to_bf16_kernel<<<cdiv(NB, T), T, 0, stream>>>(h,   hB,  (int)NB);
    f32_to_bf16_kernel<<<cdiv(NB, T), T, 0, stream>>>(hbc, hcB, (int)NB);
    f32_to_bf16_kernel<<<cdiv(NB, T), T, 0, stream>>>(hbs, hsB, (int)NB);

    dim3 gN(NT_ / BM, NF_ / BN);     // 256 x 6
    dim3 gG(NT_ / BM, G3_ / BN);     // 256 x 18

    // ---- intra-turn GAT ----
    gemm_bf16_kernel<<<gN, T, 0, stream>>>(hB, BTg, bg, Wh, NT_, NF_, NF_);
    scores_kernel<<<cdiv((size_t)NT_ * H_, T), T, 0, stream>>>(Wh, ag, scAs, scAd, NT_);
    attend_kernel<<<NT_, T, 0, stream>>>(intra_src, Wh, scAs, scAd, agb,
                                         xF, xB, 1.0f, 0);

    // ---- cross-turn inversed GAT (counter + support share Wx) ----
    gemm_bf16_kernel<<<gN, T, 0, stream>>>(xB,  BTx, bx, Wxd, NT_, NF_, NF_);
    gemm_bf16_kernel<<<gN, T, 0, stream>>>(hcB, BTx, bx, Wcs, NT_, NF_, NF_);
    gemm_bf16_kernel<<<gN, T, 0, stream>>>(hsB, BTx, bx, Wss, NT_, NF_, NF_);
    scores_kernel<<<cdiv((size_t)NT_ * H_, T), T, 0, stream>>>(Wxd, ax, scXs, scXd, NT_);
    scores_kernel<<<cdiv((size_t)NT_ * H_, T), T, 0, stream>>>(Wcs, ax, scCs, scCd, NT_);
    scores_kernel<<<cdiv((size_t)NT_ * H_, T), T, 0, stream>>>(Wss, ax, scSs, scSd, NT_);
    attend_kernel<<<NT_, T, 0, stream>>>(c_src, Wcs, scCs, scXd, axb,
                                         hidF, nullptr, 0.5f, 0);
    attend_kernel<<<NT_, T, 0, stream>>>(s_src, Wss, scSs, scXd, axb,
                                         hidF, hidB, 0.5f, 1);

    // ---- GRU gates ----
    gemm_bf16_kernel<<<gG, T, 0, stream>>>(xB,   BTih, bih, gi, NT_, G3_, NF_);
    gemm_bf16_kernel<<<gG, T, 0, stream>>>(hidB, BThh, bhh, gh, NT_, G3_, NF_);
    gru_kernel<<<cdiv(NB, T), T, 0, stream>>>(gi, gh, hidF, (float*)d_out);
}
